// StackedBiGRUModel_42185168782068
// MI455X (gfx1250) — compile-verified, matrix operand order checked by reading, not run on hardware
//
#include <hip/hip_runtime.h>

// ---------------------------------------------------------------------------
// StackedBiGRU for MI455X (gfx1250, wave32, WMMA + async-to-LDS DMA).
//   B=256, T=512, V=30000, E=256, H=32.
//   Parallel gi GEMMs   -> v_wmma_f32_16x16x32_bf16 (bf16 in, f32 accum),
//                          staging via global_load_async_to_lds_b128 (ASYNCcnt)
//   Sequential GRU scan -> v_wmma_f32_16x16x4_f32 (fp32 recurrent GEMM in LDS)
// ---------------------------------------------------------------------------

typedef __attribute__((ext_vector_type(16))) __bf16          v16bf;
typedef __attribute__((ext_vector_type(16))) unsigned short  v16u;
typedef __attribute__((ext_vector_type(8)))  float           v8f;
typedef __attribute__((ext_vector_type(2)))  float           v2f;

#define BB   256
#define TT   512
#define EE   256
#define HH   32
#define G3   96          // 3*H
#define MTOT (BB * TT)   // 131072 rows of gi
#define TILES_PER_WG 8

__device__ __forceinline__ unsigned short f2bf(float f) {
    unsigned u = __builtin_bit_cast(unsigned, f);
    unsigned r = u + 0x7FFFu + ((u >> 16) & 1u);   // round-to-nearest-even
    return (unsigned short)(r >> 16);
}

__device__ __forceinline__ unsigned lds_off(const void* p) {
    // Generic LDS pointers carry the wave-relative LDS byte offset in addr[31:0].
    return (unsigned)(unsigned long long)p;
}

__device__ __forceinline__ void async_ld_b128(unsigned lds_dst, unsigned long long gaddr) {
    asm volatile("global_load_async_to_lds_b128 %0, %1, off"
                 :: "v"(lds_dst), "v"(gaddr) : "memory");
}
__device__ __forceinline__ void wait_async0() {
    asm volatile("s_wait_asynccnt 0" ::: "memory");
}

// ---------------------------------------------------------------------------
// Prep: fp32 -> bf16 bit conversion (embed table, Wih weights).
// ---------------------------------------------------------------------------
__global__ __launch_bounds__(256)
void bf16_cast_kernel(const float* __restrict__ src, unsigned short* __restrict__ dst, int n)
{
    int i = blockIdx.x * 256 + threadIdx.x;
    if (i < n) dst[i] = f2bf(src[i]);
}

// ---------------------------------------------------------------------------
// gi = A @ Wih^T + bih. A is bf16 [rows,K], gathered through ids when
// ids != nullptr (fused embedding lookup). 192 threads = 6 waves; wave w owns
// N-tile [16w,16w+16). One WG processes TILES_PER_WG M-tiles, staging Wih once.
// All LDS staging via async-to-LDS DMA; rows padded 16B for bank spread.
// ---------------------------------------------------------------------------
template<int K>
__global__ __launch_bounds__(192)
void gi_gemm_kernel(const unsigned short* __restrict__ Abf,  // [rows,K] bf16
                    const int*            __restrict__ ids,  // [M] or nullptr
                    const unsigned short* __restrict__ Wbf,  // [96,K] bf16
                    const float*          __restrict__ bih,  // [96]
                    float*                __restrict__ gi)   // [M,96]
{
    constexpr int KS = K + 8;              // padded row stride (elements)
    constexpr int ROWB = K * 2;            // row bytes
    __shared__ unsigned short sA[16 * KS];
    __shared__ unsigned short sB[96 * KS];

    const int tid  = threadIdx.x;
    const int wave = tid >> 5, lane = tid & 31;
    const int hi   = lane >> 4, lm = lane & 15;
    const int n0   = wave * 16;

    // Stage B once per WG: one wave-wide async b128 op per weight row.
    for (int n = wave; n < G3; n += 6) {
        if (lane * 16 < ROWB) {
            unsigned long long ga =
                (unsigned long long)(size_t)Wbf + (unsigned long long)n * ROWB + lane * 16;
            async_ld_b128(lds_off(&sB[n * KS]) + lane * 16, ga);
        }
    }
    const float bias = bih[n0 + lm];

    for (int tile = 0; tile < TILES_PER_WG; ++tile) {
        const int m0 = (blockIdx.x * TILES_PER_WG + tile) * 16;

        // Stage A tile: 16 rows, gathered via token ids (embedding fuse).
        for (int r = wave; r < 16; r += 6) {
            if (lane * 16 < ROWB) {
                long row = ids ? (long)ids[m0 + r] : (long)(m0 + r);
                unsigned long long ga =
                    (unsigned long long)(size_t)Abf + (unsigned long long)row * ROWB + lane * 16;
                async_ld_b128(lds_off(&sA[r * KS]) + lane * 16, ga);
            }
        }
        wait_async0();
        __syncthreads();

        v8f c = {};
        for (int kk = 0; kk < K; kk += 32) {
            v16u au, bu;
#pragma unroll
            for (int i = 0; i < 16; ++i) {
                // 16-bit A 16x32: lane<16 holds K {0..7,16..23}, lane>=16 {8..15,24..31}
                int k = kk + hi * 8 + (i < 8 ? i : i + 8);
                au[i] = sA[lm * KS + k];
                bu[i] = sB[(n0 + lm) * KS + k];  // B[k][n] = Wih[n][k]
            }
            c = __builtin_amdgcn_wmma_f32_16x16x32_bf16(
                    false, __builtin_bit_cast(v16bf, au),
                    false, __builtin_bit_cast(v16bf, bu),
                    (short)0, c, false, false);
        }

        // C/D layout: VGPR r -> M = r (+8 for hi lanes), N = lane&15
#pragma unroll
        for (int r = 0; r < 8; ++r)
            gi[(long)(m0 + r + hi * 8) * G3 + (n0 + lm)] = c[r] + bias;
        __syncthreads();   // protect sA from next tile's async writes
    }
}

// ---------------------------------------------------------------------------
// Sequential GRU scan. One WG = 16 batch rows, full T loop. Per step:
// gh = h @ Whh^T + bhh via fp32 WMMA (6 waves x 16x16 N-tiles, 8x K4),
// VALU gates, h resident in LDS. Layer-0 writes x1 in bf16 for the L1 GEMM.
// ---------------------------------------------------------------------------
__global__ __launch_bounds__(192)
void gru_scan_kernel(const float* __restrict__ gi,       // [B*T,96], includes bih
                     const float* __restrict__ Whh,      // [96,32]
                     const float* __restrict__ bhh,      // [96]
                     const int*   __restrict__ mask,     // [B,T]
                     unsigned short* __restrict__ outbf, // [B*T,64] bf16 or nullptr
                     int out_col,                        // 0 (fwd) or 32 (bwd)
                     float*       __restrict__ hfin,     // [B,32] or nullptr
                     int reverse)
{
    __shared__ float sH[16 * 36];    // hidden state, padded rows
    __shared__ float sGH[16 * 100];  // gh = h@Whh^T + bhh, padded rows
    __shared__ float sW[96 * 36];    // Whh, padded rows
    __shared__ float sbh[G3];

    const int b0  = blockIdx.x * 16;
    const int tid = threadIdx.x;

    for (int i = tid; i < G3 * HH; i += 192) sW[(i / HH) * 36 + (i % HH)] = Whh[i];
    for (int i = tid; i < G3; i += 192)      sbh[i] = bhh[i];
    for (int i = tid; i < 16 * 36; i += 192) sH[i] = 0.0f;
    __syncthreads();

    const int wave = tid >> 5, lane = tid & 31;
    const int hi = lane >> 4, lm = lane & 15;
    const int n0 = wave * 16;

    for (int s = 0; s < TT; ++s) {
        const int t = reverse ? (TT - 1 - s) : s;

        // gh tile: C init with bhh broadcast, 8x WMMA f32 16x16x4
        v8f c;
        float bn = sbh[n0 + lm];
#pragma unroll
        for (int r = 0; r < 8; ++r) c[r] = bn;
#pragma unroll
        for (int kk = 0; kk < HH; kk += 4) {
            // f32 A 16x4: lane<16 -> K {kk,kk+1}; lane>=16 -> K {kk+2,kk+3}
            int kb = kk + hi * 2;
            v2f a, b;
            a.x = sH[lm * 36 + kb];
            a.y = sH[lm * 36 + kb + 1];
            b.x = sW[(n0 + lm) * 36 + kb];       // B[k][n] = Whh[n][k]
            b.y = sW[(n0 + lm) * 36 + kb + 1];
            c = __builtin_amdgcn_wmma_f32_16x16x4_f32(
                    false, a, false, b, (short)0, c, false, false);
        }
#pragma unroll
        for (int r = 0; r < 8; ++r)
            sGH[(r + hi * 8) * 100 + (n0 + lm)] = c[r];
        __syncthreads();

        // Prefetch next timestep's gi rows into the cache hierarchy.
        if (s + 1 < TT && tid < 16) {
            int tn = reverse ? (t - 1) : (t + 1);
            __builtin_prefetch(&gi[((long)(b0 + tid) * TT + tn) * G3], 0, 1);
        }

        // GRU gates (r,z,n order) + packed-sequence mask freeze.
        for (int e = tid; e < 16 * HH; e += 192) {
            int bb = e >> 5, j = e & 31;
            long base = ((long)(b0 + bb) * TT + t) * G3;
            float hr = sGH[bb * 100 + j];
            float hz = sGH[bb * 100 + HH + j];
            float hn = sGH[bb * 100 + 2 * HH + j];
            float ir = gi[base + j];
            float iz = gi[base + HH + j];
            float in = gi[base + 2 * HH + j];
            float r = 1.0f / (1.0f + __expf(-(ir + hr)));
            float z = 1.0f / (1.0f + __expf(-(iz + hz)));
            float n = tanhf(in + r * hn);
            float h = sH[bb * 36 + j];
            float hnew = (1.0f - z) * n + z * h;
            h = mask[(b0 + bb) * TT + t] ? hnew : h;
            sH[bb * 36 + j] = h;
            if (outbf)
                outbf[((long)(b0 + bb) * TT + t) * 64 + out_col + j] = f2bf(h);
        }
        __syncthreads();
    }

    if (hfin)
        for (int e = tid; e < 16 * HH; e += 192)
            hfin[(long)(b0 + (e >> 5)) * HH + (e & 31)] = sH[(e >> 5) * 36 + (e & 31)];
}

// ---------------------------------------------------------------------------
// out = concat(hf, hb) @ Wout^T + bout    [256,64] x [64,6]
// ---------------------------------------------------------------------------
__global__ __launch_bounds__(256)
void out_proj_kernel(const float* __restrict__ hf, const float* __restrict__ hb,
                     const float* __restrict__ Wout, const float* __restrict__ bout,
                     float* __restrict__ out)
{
    int b = blockIdx.x * blockDim.x + threadIdx.x;
    if (b >= BB) return;
    float top[64];
#pragma unroll
    for (int j = 0; j < 32; ++j) top[j] = hf[b * 32 + j];
#pragma unroll
    for (int j = 0; j < 32; ++j) top[32 + j] = hb[b * 32 + j];
#pragma unroll
    for (int o = 0; o < 6; ++o) {
        float acc = bout[o];
#pragma unroll
        for (int j = 0; j < 64; ++j) acc += top[j] * Wout[o * 64 + j];
        out[b * 6 + o] = acc;
    }
}

// ---------------------------------------------------------------------------
extern "C" void kernel_launch(void* const* d_in, const int* in_sizes, int n_in,
                              void* d_out, int out_size, void* d_ws, size_t ws_size,
                              hipStream_t stream) {
    const int*   ids   = (const int*)d_in[0];
    const int*   mask  = (const int*)d_in[1];
    const float* embed = (const float*)d_in[2];
    const float* Wih_l0f = (const float*)d_in[3];
    const float* Whh_l0f = (const float*)d_in[4];
    const float* bih_l0f = (const float*)d_in[5];
    const float* bhh_l0f = (const float*)d_in[6];
    const float* Wih_l0b = (const float*)d_in[7];
    const float* Whh_l0b = (const float*)d_in[8];
    const float* bih_l0b = (const float*)d_in[9];
    const float* bhh_l0b = (const float*)d_in[10];
    const float* Wih_l1f = (const float*)d_in[11];
    const float* Whh_l1f = (const float*)d_in[12];
    const float* bih_l1f = (const float*)d_in[13];
    const float* bhh_l1f = (const float*)d_in[14];
    const float* Wih_l1b = (const float*)d_in[15];
    const float* Whh_l1b = (const float*)d_in[16];
    const float* bih_l1b = (const float*)d_in[17];
    const float* bhh_l1b = (const float*)d_in[18];
    const float* Wout    = (const float*)d_in[19];
    const float* bout    = (const float*)d_in[20];

    // Workspace layout: fp32 gi buffers (reused by layer 1), then bf16 regions.
    char* ws = (char*)d_ws;
    float* giA = (float*)ws;                                    ws += (size_t)G3 * MTOT * 4;
    float* giB = (float*)ws;                                    ws += (size_t)G3 * MTOT * 4;
    unsigned short* x1bf  = (unsigned short*)ws;                ws += (size_t)64 * MTOT * 2;
    unsigned short* embbf = (unsigned short*)ws;                ws += (size_t)30000 * EE * 2;
    unsigned short* w0f   = (unsigned short*)ws;                ws += (size_t)G3 * EE * 2;
    unsigned short* w0b   = (unsigned short*)ws;                ws += (size_t)G3 * EE * 2;
    unsigned short* w1f   = (unsigned short*)ws;                ws += (size_t)G3 * 64 * 2;
    unsigned short* w1b   = (unsigned short*)ws;                ws += (size_t)G3 * 64 * 2;
    float* hf1 = (float*)ws;                                    ws += (size_t)BB * HH * 4;
    float* hb1 = (float*)ws;

    // Prep: one-time fp32 -> bf16 conversion of embed table and Wih weights.
    bf16_cast_kernel<<<(30000 * EE + 255) / 256, 256, 0, stream>>>(embed, embbf, 30000 * EE);
    bf16_cast_kernel<<<(G3 * EE + 255) / 256, 256, 0, stream>>>(Wih_l0f, w0f, G3 * EE);
    bf16_cast_kernel<<<(G3 * EE + 255) / 256, 256, 0, stream>>>(Wih_l0b, w0b, G3 * EE);
    bf16_cast_kernel<<<(G3 * 64 + 255) / 256, 256, 0, stream>>>(Wih_l1f, w1f, G3 * 64);
    bf16_cast_kernel<<<(G3 * 64 + 255) / 256, 256, 0, stream>>>(Wih_l1b, w1b, G3 * 64);

    const int gemmBlocks = MTOT / (16 * TILES_PER_WG);   // 1024
    const int scanBlocks = BB / 16;                      // 16

    // Layer 0: fused embedding gather + input-gate GEMMs (bf16 WMMA).
    gi_gemm_kernel<EE><<<gemmBlocks, 192, 0, stream>>>(embbf, ids, w0f, bih_l0f, giA);
    gi_gemm_kernel<EE><<<gemmBlocks, 192, 0, stream>>>(embbf, ids, w0b, bih_l0b, giB);
    // Layer 0 scans: write x1 halves directly in bf16.
    gru_scan_kernel<<<scanBlocks, 192, 0, stream>>>(giA, Whh_l0f, bhh_l0f, mask, x1bf, 0,  nullptr, 0);
    gru_scan_kernel<<<scanBlocks, 192, 0, stream>>>(giB, Whh_l0b, bhh_l0b, mask, x1bf, 32, nullptr, 1);
    // Layer 1 input-gate GEMMs from bf16 x1 (K=64), reusing gi buffers.
    gi_gemm_kernel<64><<<gemmBlocks, 192, 0, stream>>>(x1bf, nullptr, w1f, bih_l1f, giA);
    gi_gemm_kernel<64><<<gemmBlocks, 192, 0, stream>>>(x1bf, nullptr, w1b, bih_l1b, giB);
    // Layer 1 scans: only final hidden states needed.
    gru_scan_kernel<<<scanBlocks, 192, 0, stream>>>(giA, Whh_l1f, bhh_l1f, mask, nullptr, 0, hf1, 0);
    gru_scan_kernel<<<scanBlocks, 192, 0, stream>>>(giB, Whh_l1b, bhh_l1b, mask, nullptr, 0, hb1, 1);
    // Final projection [B,6].
    out_proj_kernel<<<1, 256, 0, stream>>>(hf1, hb1, Wout, bout, (float*)d_out);
}
